// CrossAttention_71863392797502
// MI455X (gfx1250) — matrix operand order repaired and checked by reading, not compiled, not run on hardware
//
#include <hip/hip_runtime.h>

// ---- problem constants (match reference) ----
#define B_    4
#define NQ_   1024
#define NKV_  2048
#define DM    1024      // Q_DIM = K_DIM = V_DIM = INNER
#define HEADS_ 16
#define HD_    64       // DIM_HEAD
#define INNER_ 1024

typedef __bf16 v8bf  __attribute__((ext_vector_type(8)));
typedef __bf16 v16bf __attribute__((ext_vector_type(16)));
typedef float  v8f   __attribute__((ext_vector_type(8)));

#define SHUF16(lo, hi) __builtin_shufflevector((lo),(hi),0,1,2,3,4,5,6,7,8,9,10,11,12,13,14,15)

__device__ __forceinline__ v8f wmma_bf16(v16bf a, v16bf b, v8f c) {
  return __builtin_amdgcn_wmma_f32_16x16x32_bf16(false, a, false, b, (short)0, c, false, false);
}

// CDNA5 async copy: global -> LDS, 16B per lane, tracked by ASYNCcnt.
__device__ __forceinline__ void async_copy_b128(void* lds_dst, const void* gsrc) {
  unsigned l = (unsigned)(unsigned long long)lds_dst;
  asm volatile("global_load_async_to_lds_b128 %0, %1, off" :: "v"(l), "v"(gsrc) : "memory");
}
__device__ __forceinline__ void wait_asynccnt0() {
#if __has_builtin(__builtin_amdgcn_s_wait_asynccnt)
  __builtin_amdgcn_s_wait_asynccnt(0);
#else
  asm volatile("s_wait_asynccnt 0x0" ::: "memory");
#endif
}

// ---------------------------------------------------------------------------
// LayerNorm (fp32 in) -> bf16 out.  One 256-thread WG per row.
// ---------------------------------------------------------------------------
__global__ __launch_bounds__(256) void k_layernorm_bf16(
    const float* __restrict__ x, const float* __restrict__ gamma,
    const float* __restrict__ beta, __bf16* __restrict__ xn) {
  int row = blockIdx.x;
  int tid = threadIdx.x;
  __shared__ float2 red[256];
  const float* xr = x + (size_t)row * DM;
  float v[4];
  float s = 0.f, s2 = 0.f;
#pragma unroll
  for (int i = 0; i < 4; ++i) {
    v[i] = xr[tid + i * 256];
    s += v[i];
    s2 += v[i] * v[i];
  }
  red[tid] = make_float2(s, s2);
  __syncthreads();
  for (int off = 128; off > 0; off >>= 1) {
    if (tid < off) {
      red[tid].x += red[tid + off].x;
      red[tid].y += red[tid + off].y;
    }
    __syncthreads();
  }
  float mu   = red[0].x * (1.f / DM);
  float var  = red[0].y * (1.f / DM) - mu * mu;
  float rstd = rsqrtf(var + 1e-5f);
  __bf16* orow = xn + (size_t)row * DM;
#pragma unroll
  for (int i = 0; i < 4; ++i) {
    int c = tid + i * 256;
    orow[c] = (__bf16)((v[i] - mu) * rstd * gamma[c] + beta[c]);
  }
}

// ---------------------------------------------------------------------------
// Transpose-convert: in[R][C] fp32 -> out[C][R] bf16.  LDS-tiled 32x32.
// ---------------------------------------------------------------------------
__global__ void k_transpose_bf16(const float* __restrict__ in,
                                 __bf16* __restrict__ out, int R, int C) {
  __shared__ float t[32][33];
  int c0 = blockIdx.x * 32, r0 = blockIdx.y * 32;
  int xx = threadIdx.x, yy = threadIdx.y;  // (32, 8)
#pragma unroll
  for (int i = 0; i < 32; i += 8)
    t[yy + i][xx] = in[(size_t)(r0 + yy + i) * C + c0 + xx];
  __syncthreads();
#pragma unroll
  for (int i = 0; i < 32; i += 8)
    out[(size_t)(c0 + yy + i) * R + r0 + xx] = (__bf16)t[xx][yy + i];
}

// ---------------------------------------------------------------------------
// GEMM shared pieces.  Block tile 128x128x32, 8 waves, wave tile 64x32.
// ---------------------------------------------------------------------------
union SmemG {
  struct { __bf16 A[2][128][40]; __bf16 B[2][128][40]; } t;  // 40960 B
  __bf16 cb[128][136];                                       // 34816 B (epilogue)
  float  cf[64][132];                                        // 33792 B (epilogue, half tile)
};

__device__ __forceinline__ void tile_mma(const __bf16 (*Asb)[40], const __bf16 (*Bsb)[40],
                                         v8f acc[4][2], int wm, int wn, int tl, int th) {
  v16bf bfr[2];
#pragma unroll
  for (int j = 0; j < 2; ++j) {
    int n = wn * 32 + j * 16 + tl;
    v8bf lo = *(const v8bf*)&Bsb[n][th * 8];
    v8bf hi = *(const v8bf*)&Bsb[n][16 + th * 8];
    bfr[j] = SHUF16(lo, hi);
  }
#pragma unroll
  for (int i = 0; i < 4; ++i) {
    int r = wm * 64 + i * 16 + tl;
    v8bf lo = *(const v8bf*)&Asb[r][th * 8];
    v8bf hi = *(const v8bf*)&Asb[r][16 + th * 8];
    v16bf afr = SHUF16(lo, hi);
#pragma unroll
    for (int j = 0; j < 2; ++j)
      acc[i][j] = wmma_bf16(afr, bfr[j], acc[i][j]);
  }
}

// Coalesced bf16 epilogue: scatter C-layout into LDS, then linear b128 stores.
__device__ __forceinline__ void store_c_bf16(__bf16* __restrict__ C, v8f acc[4][2],
                                             SmemG* sm, int m0, int n0, int N,
                                             int wm, int wn, int tl, int th, int tid) {
  __syncthreads();  // all waves done reading tile buffers
#pragma unroll
  for (int i = 0; i < 4; ++i)
#pragma unroll
    for (int j = 0; j < 2; ++j)
#pragma unroll
      for (int r = 0; r < 8; ++r)
        sm->cb[wm * 64 + i * 16 + r + th * 8][wn * 32 + j * 16 + tl] = (__bf16)acc[i][j][r];
  __syncthreads();
  int row = tid >> 1, colh = (tid & 1) * 64;
#pragma unroll
  for (int q = 0; q < 8; ++q) {
    v8bf v = *(const v8bf*)&sm->cb[row][colh + q * 8];
    *(v8bf*)&C[(size_t)(m0 + row) * N + n0 + colh + q * 8] = v;
  }
}

// Coalesced f32 epilogue: two 64-row passes so the repack buffer fits in LDS.
__device__ __forceinline__ void store_c_f32(float* __restrict__ C, v8f acc[4][2],
                                            SmemG* sm, int m0, int n0, int N,
                                            int wm, int wn, int tl, int th, int tid) {
#pragma unroll
  for (int hh = 0; hh < 2; ++hh) {
    __syncthreads();
    if (wm == hh) {
#pragma unroll
      for (int i = 0; i < 4; ++i)
#pragma unroll
        for (int j = 0; j < 2; ++j)
#pragma unroll
        for (int r = 0; r < 8; ++r)
          sm->cf[i * 16 + r + th * 8][wn * 32 + j * 16 + tl] = acc[i][j][r];
    }
    __syncthreads();
    int row = tid >> 2, seg = (tid & 3) * 32;
#pragma unroll
    for (int q = 0; q < 8; ++q) {
      float4 v = *(const float4*)&sm->cf[row][seg + q * 4];
      *(float4*)&C[(size_t)(m0 + hh * 64 + row) * N + n0 + seg + q * 4] = v;
    }
  }
}

// ---------------------------------------------------------------------------
// bf16 GEMM, double-buffered via CDNA5 async global->LDS DMA.
// C[M][N] = A[M][K] * Bt[N][K]^T.
// ---------------------------------------------------------------------------
template <bool OUT_BF16>
__global__ __launch_bounds__(256) void k_gemm_bf16_async(
    const __bf16* __restrict__ A, const __bf16* __restrict__ Bt,
    void* __restrict__ Cout, int N, int K) {
  __shared__ alignas(16) SmemG sm;
  int tid = threadIdx.x, lane = tid & 31, wave = tid >> 5;
  int wm = wave & 1, wn = wave >> 1;
  int tl = lane & 15, th = lane >> 4;
  int m0 = blockIdx.y * 128, n0 = blockIdx.x * 128;
  int ldr = tid >> 1, ldh = (tid & 1) * 16;
  v8f acc[4][2] = {};
  int steps = K / 32;

  auto stage = [&](int step, int buf) {
    const __bf16* pa = A + (size_t)(m0 + ldr) * K + step * 32 + ldh;
    async_copy_b128(&sm.t.A[buf][ldr][ldh], pa);
    async_copy_b128(&sm.t.A[buf][ldr][ldh + 8], pa + 8);
    const __bf16* pb = Bt + (size_t)(n0 + ldr) * K + step * 32 + ldh;
    async_copy_b128(&sm.t.B[buf][ldr][ldh], pb);
    async_copy_b128(&sm.t.B[buf][ldr][ldh + 8], pb + 8);
  };

  stage(0, 0);
  for (int i = 0; i < steps; ++i) {
    int buf = i & 1;
    wait_asynccnt0();
    __syncthreads();
    if (i + 1 < steps) stage(i + 1, buf ^ 1);  // DMA overlaps compute below
    tile_mma(sm.t.A[buf], sm.t.B[buf], acc, wm, wn, tl, th);
  }
  if (OUT_BF16)
    store_c_bf16((__bf16*)Cout, acc, &sm, m0, n0, N, wm, wn, tl, th, tid);
  else
    store_c_f32((float*)Cout, acc, &sm, m0, n0, N, wm, wn, tl, th, tid);
}

// ---------------------------------------------------------------------------
// fp32-A GEMM (conversion fused into LDS staging), register double-buffered.
// ---------------------------------------------------------------------------
__global__ __launch_bounds__(256) void k_gemm_f32a(
    const float* __restrict__ A, const __bf16* __restrict__ Bt,
    __bf16* __restrict__ Cout, int N, int K) {
  __shared__ alignas(16) SmemG sm;
  int tid = threadIdx.x, lane = tid & 31, wave = tid >> 5;
  int wm = wave & 1, wn = wave >> 1;
  int tl = lane & 15, th = lane >> 4;
  int m0 = blockIdx.y * 128, n0 = blockIdx.x * 128;
  int ldr = tid >> 1, ldh = (tid & 1) * 16;
  v8f acc[4][2] = {};
  int steps = K / 32;

  float4 ar[4];
  v8bf br[2];
  auto load_regs = [&](int step) {
    const float4* pa = (const float4*)(A + (size_t)(m0 + ldr) * K + step * 32 + ldh);
#pragma unroll
    for (int q = 0; q < 4; ++q) ar[q] = pa[q];
    const v8bf* pb = (const v8bf*)(Bt + (size_t)(n0 + ldr) * K + step * 32 + ldh);
    br[0] = pb[0];
    br[1] = pb[1];
  };
  auto store_lds = [&](int buf) {
    v8bf lo, hi;
    lo[0] = (__bf16)ar[0].x; lo[1] = (__bf16)ar[0].y; lo[2] = (__bf16)ar[0].z; lo[3] = (__bf16)ar[0].w;
    lo[4] = (__bf16)ar[1].x; lo[5] = (__bf16)ar[1].y; lo[6] = (__bf16)ar[1].z; lo[7] = (__bf16)ar[1].w;
    hi[0] = (__bf16)ar[2].x; hi[1] = (__bf16)ar[2].y; hi[2] = (__bf16)ar[2].z; hi[3] = (__bf16)ar[2].w;
    hi[4] = (__bf16)ar[3].x; hi[5] = (__bf16)ar[3].y; hi[6] = (__bf16)ar[3].z; hi[7] = (__bf16)ar[3].w;
    *(v8bf*)&sm.t.A[buf][ldr][ldh]     = lo;
    *(v8bf*)&sm.t.A[buf][ldr][ldh + 8] = hi;
    *(v8bf*)&sm.t.B[buf][ldr][ldh]     = br[0];
    *(v8bf*)&sm.t.B[buf][ldr][ldh + 8] = br[1];
  };

  load_regs(0);
  store_lds(0);
  for (int i = 0; i < steps; ++i) {
    int buf = i & 1;
    __syncthreads();
    if (i + 1 < steps) load_regs(i + 1);  // global latency overlaps compute
    tile_mma(sm.t.A[buf], sm.t.B[buf], acc, wm, wn, tl, th);
    if (i + 1 < steps) store_lds(buf ^ 1);
  }
  store_c_bf16(Cout, acc, &sm, m0, n0, N, wm, wn, tl, th, tid);
}

// ---------------------------------------------------------------------------
// Flash attention: WG = (b, h, 128 q rows); 8 waves, 16 q rows per wave.
// ---------------------------------------------------------------------------
union SmemA {
  struct {
    __bf16 Ks[32][72];     // [kv][d]
    __bf16 Vs[64][40];     // [d][kv] transposed
    __bf16 Ps[8][16][40];  // per-wave P tile
  } t;                     // 19968 B
  __bf16 cb[128][72];      // 18432 B (epilogue repack)
};

__global__ __launch_bounds__(256) void k_attention(
    const __bf16* __restrict__ Qp, const __bf16* __restrict__ Kp,
    const __bf16* __restrict__ Vp, const unsigned char* __restrict__ mask,
    __bf16* __restrict__ AO) {
  __shared__ alignas(16) SmemA sm;

  int tid = threadIdx.x, lane = tid & 31, wave = tid >> 5;
  int tl = lane & 15, th = lane >> 4;
  int wg = blockIdx.x;
  int qtile = wg & 7;           // NQ/128 = 8
  int h = (wg >> 3) & 15;       // HEADS = 16
  int b = wg >> 7;
  int q0 = qtile * 128;
  int qw = q0 + wave * 16;

  // Q fragments (A layout), resident for the whole kv loop.
  v16bf qf[2];
  {
    const __bf16* qb = Qp + (size_t)(b * NQ_ + qw + tl) * INNER_ + h * HD_;
#pragma unroll
    for (int d = 0; d < 2; ++d) {
      v8bf lo = *(const v8bf*)(qb + d * 32 + th * 8);
      v8bf hi = *(const v8bf*)(qb + d * 32 + 16 + th * 8);
      qf[d] = SHUF16(lo, hi);
    }
  }

  float mrow[8], lsum[8];
  v8f oacc[4] = {};
#pragma unroll
  for (int r = 0; r < 8; ++r) { mrow[r] = -3.4028235e38f; lsum[r] = 0.f; }
  const float scale = 0.125f;  // 64^-0.5

  int kvr = tid >> 3, dc = (tid & 7) * 8;  // cooperative tile-loader coords

  for (int kv0 = 0; kv0 < NKV_; kv0 += 32) {
    size_t src = (size_t)(b * NKV_ + kv0 + kvr) * INNER_ + h * HD_ + dc;
    async_copy_b128(&sm.t.Ks[kvr][dc], Kp + src);  // K tile via async DMA
    v8bf vv = *(const v8bf*)(Vp + src);            // V tile: load + transpose
#pragma unroll
    for (int e = 0; e < 8; ++e) sm.t.Vs[dc + e][kvr] = vv[e];
    wait_asynccnt0();
    __syncthreads();

    // S = Q K^T : 2 n-frags x 2 d-steps
    v8f s[2] = {};
#pragma unroll
    for (int d = 0; d < 2; ++d)
#pragma unroll
      for (int j = 0; j < 2; ++j) {
        int n = j * 16 + tl;  // kv column
        v8bf lo = *(const v8bf*)&sm.t.Ks[n][d * 32 + th * 8];
        v8bf hi = *(const v8bf*)&sm.t.Ks[n][d * 32 + 16 + th * 8];
        v16bf bk = SHUF16(lo, hi);
        s[j] = wmma_bf16(qf[d], bk, s[j]);
      }

    // scale + mask (mask nonzero -> -FLT_MAX, matching reference)
#pragma unroll
    for (int j = 0; j < 2; ++j)
#pragma unroll
      for (int r = 0; r < 8; ++r) {
        int qrow = qw + r + th * 8;
        int kvc = kv0 + j * 16 + tl;
        float sv = s[j][r] * scale;
        if (mask[(size_t)(b * NQ_ + qrow) * NKV_ + kvc]) sv = -3.4028235e38f;
        s[j][r] = sv;
      }

    // online softmax update (row reductions stay inside 16-lane halves)
    float pr[2][8];
#pragma unroll
    for (int r = 0; r < 8; ++r) {
      float mx = fmaxf(s[0][r], s[1][r]);
      for (int m = 8; m > 0; m >>= 1) mx = fmaxf(mx, __shfl_xor(mx, m, 32));
      float mn = fmaxf(mrow[r], mx);
      float corr = __expf(mrow[r] - mn);
      float p0 = __expf(s[0][r] - mn);
      float p1 = __expf(s[1][r] - mn);
      float sum = p0 + p1;
      for (int m = 8; m > 0; m >>= 1) sum += __shfl_xor(sum, m, 32);
      lsum[r] = lsum[r] * corr + sum;
      mrow[r] = mn;
      pr[0][r] = p0; pr[1][r] = p1;
#pragma unroll
      for (int f = 0; f < 4; ++f) oacc[f][r] = oacc[f][r] * corr;
    }

    // P (C layout) -> LDS -> re-read in A-fragment layout
#pragma unroll
    for (int j = 0; j < 2; ++j)
#pragma unroll
      for (int r = 0; r < 8; ++r)
        sm.t.Ps[wave][r + th * 8][j * 16 + tl] = (__bf16)pr[j][r];
    __syncthreads();

    v16bf pa;
    {
      v8bf lo = *(const v8bf*)&sm.t.Ps[wave][tl][th * 8];
      v8bf hi = *(const v8bf*)&sm.t.Ps[wave][tl][16 + th * 8];
      pa = SHUF16(lo, hi);
    }
#pragma unroll
    for (int f = 0; f < 4; ++f) {
      int dcol = f * 16 + tl;
      v8bf lo = *(const v8bf*)&sm.t.Vs[dcol][th * 8];
      v8bf hi = *(const v8bf*)&sm.t.Vs[dcol][16 + th * 8];
      v16bf bv = SHUF16(lo, hi);
      oacc[f] = wmma_bf16(pa, bv, oacc[f]);
    }
    __syncthreads();  // also guards next iteration's tile overwrite + epilogue
  }

  // epilogue: normalize, repack via LDS, coalesced b128 stores
#pragma unroll
  for (int r = 0; r < 8; ++r) {
    float il = 1.f / lsum[r];
#pragma unroll
    for (int f = 0; f < 4; ++f)
      sm.cb[wave * 16 + r + th * 8][f * 16 + tl] = (__bf16)(oacc[f][r] * il);
  }
  __syncthreads();
  int row = tid >> 1, ch = (tid & 1) * 32;
#pragma unroll
  for (int q = 0; q < 4; ++q) {
    v8bf v = *(const v8bf*)&sm.cb[row][ch + q * 8];
    *(v8bf*)&AO[(size_t)(b * NQ_ + q0 + row) * INNER_ + h * HD_ + ch + q * 8] = v;
  }
}

// ---------------------------------------------------------------------------
// host driver
// ---------------------------------------------------------------------------
extern "C" void kernel_launch(void* const* d_in, const int* in_sizes, int n_in,
                              void* d_out, int out_size, void* d_ws, size_t ws_size,
                              hipStream_t stream) {
  (void)in_sizes; (void)n_in; (void)out_size; (void)ws_size;
  const float* x     = (const float*)d_in[0];
  const float* key_t = (const float*)d_in[1];
  const float* value = (const float*)d_in[2];
  const unsigned char* mask = (const unsigned char*)d_in[3];
  const float* Wq = (const float*)d_in[4];
  const float* Wk = (const float*)d_in[5];
  const float* Wv = (const float*)d_in[6];
  const float* Wo = (const float*)d_in[7];
  const float* gamma = (const float*)d_in[8];
  const float* beta  = (const float*)d_in[9];
  float* out = (float*)d_out;

  char* ws = (char*)d_ws;
  size_t off = 0;
  __bf16* xn  = (__bf16*)(ws + off); off += (size_t)B_ * NQ_  * DM * 2;   // 8 MB
  __bf16* Wqt = (__bf16*)(ws + off); off += (size_t)DM * INNER_ * 2;      // 2 MB
  __bf16* Wkt = (__bf16*)(ws + off); off += (size_t)DM * INNER_ * 2;
  __bf16* Wvt = (__bf16*)(ws + off); off += (size_t)DM * INNER_ * 2;
  __bf16* Wot = (__bf16*)(ws + off); off += (size_t)INNER_ * DM * 2;
  __bf16* Qb  = (__bf16*)(ws + off); off += (size_t)B_ * NQ_  * INNER_ * 2;
  __bf16* Kb  = (__bf16*)(ws + off); off += (size_t)B_ * NKV_ * INNER_ * 2;
  __bf16* Vb  = (__bf16*)(ws + off); off += (size_t)B_ * NKV_ * INNER_ * 2;
  __bf16* AO  = (__bf16*)(ws + off); off += (size_t)B_ * NQ_  * INNER_ * 2;

  // 1) LayerNorm -> bf16
  k_layernorm_bf16<<<B_ * NQ_, 256, 0, stream>>>(x, gamma, beta, xn);

  // 2) transpose-convert weights to [N][K] bf16
  dim3 tb(32, 8), tg(INNER_ / 32, DM / 32);
  k_transpose_bf16<<<tg, tb, 0, stream>>>(Wq, Wqt, DM, INNER_);
  k_transpose_bf16<<<tg, tb, 0, stream>>>(Wk, Wkt, DM, INNER_);
  k_transpose_bf16<<<tg, tb, 0, stream>>>(Wv, Wvt, DM, INNER_);
  k_transpose_bf16<<<tg, tb, 0, stream>>>(Wo, Wot, INNER_, DM);

  // 3) projections: Q from bf16 xn (async path), K/V fused fp32->bf16 path
  k_gemm_bf16_async<true><<<dim3(INNER_ / 128, (B_ * NQ_) / 128), 256, 0, stream>>>(
      xn, Wqt, Qb, INNER_, DM);
  k_gemm_f32a<<<dim3(INNER_ / 128, (B_ * NKV_) / 128), 256, 0, stream>>>(
      key_t, Wkt, Kb, INNER_, DM);
  k_gemm_f32a<<<dim3(INNER_ / 128, (B_ * NKV_) / 128), 256, 0, stream>>>(
      value, Wvt, Vb, INNER_, DM);

  // 4) fused flash attention over all (b, h, q-tile)
  k_attention<<<B_ * HEADS_ * (NQ_ / 128), 256, 0, stream>>>(Qb, Kb, Vb, mask, AO);

  // 5) output projection (fp32 out, async path)
  k_gemm_bf16_async<false><<<dim3(DM / 128, (B_ * NQ_) / 128), 256, 0, stream>>>(
      AO, Wot, out, DM, INNER_);
}
